// CrossLayerTranscoder_48541720379476
// MI455X (gfx1250) — matrix operand order repaired and checked by reading, not compiled, not run on hardware
//
#include <hip/hip_runtime.h>

// ---------------------------------------------------------------------------
// Cross-layer transcoder forward for MI455X (gfx1250, wave32, WMMA).
// - GEMMs on v_wmma_f32_16x16x32_f16 (fp32 accumulate), operands converted
//   fp32->f16 while staging through LDS.
// - 256x128 block tile, 8 waves in 4x2 grid, 64x64 per wave (4x4 WMMA tiles):
//   16 WMMAs per 16 ds_load_b128 per K-step.
// - Double-buffered LDS with register staging: global loads for chunk k+1 are
//   in flight while WMMAs consume chunk k; one barrier per K-step.
// - Decoder segment-sum folded into a flattened K loop over 1-3 sources.
// ---------------------------------------------------------------------------

#define L_LAYERS 8
#define B_TOK    1024
#define D_DIM    1024
#define F_DIM    8192

#define TMB  256   // block tile rows (M)
#define TNB  128   // block tile cols (N)
#define TKC  32    // K chunk per pipeline step
#define LDSK 40    // padded LDS stride in halves (80B rows: conflict-free)

typedef __attribute__((ext_vector_type(4)))  _Float16 v4h;
typedef __attribute__((ext_vector_type(8)))  _Float16 v8h;
typedef __attribute__((ext_vector_type(16))) _Float16 v16h;
typedef __attribute__((ext_vector_type(8)))  float    v8f;

// Per-thread register staging for one K chunk: A 256x32 (8 float4/thread),
// B 128x32 (4 float4/thread) with a 256-thread block.
struct StageRegs {
  float4 a[8];
  float4 b[4];
};

__device__ __forceinline__ void prefetch_chunk(StageRegs& r,
                                               const float* __restrict__ A,
                                               const float* __restrict__ B,
                                               int ldA, int ldB, int k0) {
#pragma unroll
  for (int i = 0; i < 8; ++i) {
    const int idx = threadIdx.x + i * 256;        // 0..2047
    const int row = idx >> 3;                     // 8 float4 per 32-wide row
    const int c4  = (idx & 7) << 2;
    r.a[i] = *(const float4*)(A + (size_t)row * ldA + k0 + c4);
  }
#pragma unroll
  for (int i = 0; i < 4; ++i) {
    const int idx = threadIdx.x + i * 256;        // 0..1023
    const int row = idx >> 3;
    const int c4  = (idx & 7) << 2;
    r.b[i] = *(const float4*)(B + (size_t)row * ldB + k0 + c4);
  }
}

__device__ __forceinline__ void commit_chunk(const StageRegs& r,
                                             _Float16* __restrict__ sA,
                                             _Float16* __restrict__ sB) {
#pragma unroll
  for (int i = 0; i < 8; ++i) {
    const int idx = threadIdx.x + i * 256;
    const int row = idx >> 3;
    const int c4  = (idx & 7) << 2;
    const float4 v = r.a[i];
    v4h h = { (_Float16)v.x, (_Float16)v.y, (_Float16)v.z, (_Float16)v.w };
    *(v4h*)(sA + row * LDSK + c4) = h;
  }
#pragma unroll
  for (int i = 0; i < 4; ++i) {
    const int idx = threadIdx.x + i * 256;
    const int row = idx >> 3;
    const int c4  = (idx & 7) << 2;
    const float4 v = r.b[i];
    v4h h = { (_Float16)v.x, (_Float16)v.y, (_Float16)v.z, (_Float16)v.w };
    *(v4h*)(sB + row * LDSK + c4) = h;
  }
}

// A fragment (16x32 MxK): lanes 0-15 -> rows; halves 0..7 = K {0..7|8..15},
// halves 8..15 = K {16..23|24..31} by lane group (ISA 7.12.2).
__device__ __forceinline__ v16h load_frag_a(const _Float16* sm, int row0) {
  const int lane = threadIdx.x & 31;
  const _Float16* p = sm + (row0 + (lane & 15)) * LDSK + ((lane >> 4) << 3);
  union { v16h v; v8h h[2]; } f;
  f.h[0] = *(const v8h*)(p);
  f.h[1] = *(const v8h*)(p + 16);
  return f.v;
}

// B fragment (32x16 KxN): lanes 0-15 -> cols; halves 0..7 = K {0..7|8..15},
// halves 8..15 = K {16..23|24..31} by lane group (same as A).
__device__ __forceinline__ v16h load_frag_b(const _Float16* sm, int col0) {
  const int lane = threadIdx.x & 31;
  const _Float16* p = sm + (col0 + (lane & 15)) * LDSK + ((lane >> 4) << 3);
  union { v16h v; v8h h[2]; } f;
  f.h[0] = *(const v8h*)(p);
  f.h[1] = *(const v8h*)(p + 16);
  return f.v;
}

// One K=32 step for this wave's 64x64 sub-tile: 16 WMMAs.
__device__ __forceinline__ void mma_step(const _Float16* sA, const _Float16* sB,
                                         v8f (&acc)[4][4], int wm, int wn) {
  v16h a[4], b[4];
#pragma unroll
  for (int mi = 0; mi < 4; ++mi) a[mi] = load_frag_a(sA, wm * 64 + mi * 16);
#pragma unroll
  for (int ni = 0; ni < 4; ++ni) b[ni] = load_frag_b(sB, wn * 64 + ni * 16);
#pragma unroll
  for (int mi = 0; mi < 4; ++mi)
#pragma unroll
    for (int ni = 0; ni < 4; ++ni)
      acc[mi][ni] = __builtin_amdgcn_wmma_f32_16x16x32_f16(
          false, a[mi], false, b[ni], (short)0, acc[mi][ni], false, false);
}

__device__ __forceinline__ float block_reduce(float v, float* red) {
  red[threadIdx.x] = v;
  __syncthreads();
  for (int s = 128; s > 0; s >>= 1) {
    if ((int)threadIdx.x < s) red[threadIdx.x] += red[threadIdx.x + s];
    __syncthreads();
  }
  return red[0];
}

// ---------------------------------------------------------------------------
// Encoder: feats[l] = relu(residual[l] @ enc_w[l]^T + enc_b[l])
// Fused L1 partial sums and per-layer L0 counts.
// ---------------------------------------------------------------------------
__global__ __launch_bounds__(256) void clt_encoder(
    const float* __restrict__ res, const float* __restrict__ enc_w,
    const float* __restrict__ enc_b, float* __restrict__ feats,
    float* __restrict__ ws_l1, float* __restrict__ ws_l0) {
  __shared__ _Float16 sA[2][TMB * LDSK];
  __shared__ _Float16 sB[2][TNB * LDSK];
  __shared__ float    red[256];

  const int l    = blockIdx.z;
  const int b0   = blockIdx.y * TMB;
  const int f0   = blockIdx.x * TNB;
  const int wave = threadIdx.x >> 5;
  const int wm   = wave & 3;     // 4 waves along M (4*64 = 256)
  const int wn   = wave >> 2;    // 2 waves along N (2*64 = 128)
  const int lane = threadIdx.x & 31;

  const float* A = res   + (size_t)l * B_TOK * D_DIM + (size_t)b0 * D_DIM;
  const float* W = enc_w + (size_t)l * F_DIM * D_DIM + (size_t)f0 * D_DIM;

  v8f zero = {};
  v8f acc[4][4];
#pragma unroll
  for (int mi = 0; mi < 4; ++mi)
#pragma unroll
    for (int ni = 0; ni < 4; ++ni) acc[mi][ni] = zero;

  StageRegs r;
  prefetch_chunk(r, A, W, D_DIM, D_DIM, 0);
  commit_chunk(r, sA[0], sB[0]);
  __syncthreads();

  const int nsteps = D_DIM / TKC;  // 32
  for (int s = 0; s < nsteps; ++s) {
    const int cur = s & 1;
    if (s + 1 < nsteps)
      prefetch_chunk(r, A, W, D_DIM, D_DIM, (s + 1) * TKC);  // in flight...
    mma_step(sA[cur], sB[cur], acc, wm, wn);                  // ...during WMMAs
    if (s + 1 < nsteps)
      commit_chunk(r, sA[cur ^ 1], sB[cur ^ 1]);
    __syncthreads();
  }

  // Epilogue: bias + relu, store feats, accumulate L1 / L0.
  float l1 = 0.f, cnt = 0.f;
  const int nlane = lane & 15;
  const int mbase = (lane >> 4) << 3;
  float* out = feats + (size_t)l * B_TOK * F_DIM;
#pragma unroll
  for (int ni = 0; ni < 4; ++ni) {
    const int   fg   = f0 + wn * 64 + ni * 16 + nlane;
    const float bias = enc_b[(size_t)l * F_DIM + fg];
#pragma unroll
    for (int mi = 0; mi < 4; ++mi) {
      const int mg0 = b0 + wm * 64 + mi * 16 + mbase;
#pragma unroll
      for (int i = 0; i < 8; ++i) {
        float v = acc[mi][ni][i] + bias;
        v = v > 0.f ? v : 0.f;
        out[(size_t)(mg0 + i) * F_DIM + fg] = v;
        l1 += v;
        cnt += (v > 0.f) ? 1.f : 0.f;
      }
    }
  }

  float tot = block_reduce(l1, red);
  if (threadIdx.x == 0) atomicAdd(ws_l1, tot);
  __syncthreads();
  tot = block_reduce(cnt, red);
  if (threadIdx.x == 0) atomicAdd(&ws_l0[l], tot);
}

// ---------------------------------------------------------------------------
// Decoder: recons[t] = sum_{i in src(t)} feats[i] @ dec_w[p(i,t)]^T
// Flattened K loop over sources keeps the LDS pipeline full across pairs.
// ---------------------------------------------------------------------------
__global__ __launch_bounds__(256) void clt_decoder(
    const float* __restrict__ feats, const float* __restrict__ dec_w,
    float* __restrict__ recons) {
  __shared__ _Float16 sA[2][TMB * LDSK];
  __shared__ _Float16 sB[2][TNB * LDSK];

  const int t    = blockIdx.z;
  const int b0   = blockIdx.y * TMB;
  const int d0   = blockIdx.x * TNB;
  const int wave = threadIdx.x >> 5;
  const int wm   = wave & 3;
  const int wn   = wave >> 2;
  const int lane = threadIdx.x & 31;

  v8f zero = {};
  v8f acc[4][4];
#pragma unroll
  for (int mi = 0; mi < 4; ++mi)
#pragma unroll
    for (int ni = 0; ni < 4; ++ni) acc[mi][ni] = zero;

  // Source pairs for target t: i in [max(0,t-2), t], pair p = prefix(i)+(t-i).
  const int i0   = (t - 2 < 0) ? 0 : t - 2;
  const int nsrc = t - i0 + 1;  // 1..3 (uniform per block)
  const float* Ap0 = feats + (size_t)(i0 + 0) * B_TOK * F_DIM + (size_t)b0 * F_DIM;
  const float* Ap1 = feats + (size_t)((i0 + 1 < L_LAYERS) ? i0 + 1 : i0) * B_TOK * F_DIM + (size_t)b0 * F_DIM;
  const float* Ap2 = feats + (size_t)((i0 + 2 < L_LAYERS) ? i0 + 2 : i0) * B_TOK * F_DIM + (size_t)b0 * F_DIM;
  const int p0 = (((i0 + 0) <= 6) ? 3 * (i0 + 0) : 20) + (t - (i0 + 0));
  const int p1 = (((i0 + 1) <= 6) ? 3 * (i0 + 1) : 20) + ((t - (i0 + 1) >= 0) ? t - (i0 + 1) : 0);
  const int p2 = (((i0 + 2) <= 6) ? 3 * (i0 + 2) : 20) + ((t - (i0 + 2) >= 0) ? t - (i0 + 2) : 0);
  const float* Wp0 = dec_w + (size_t)p0 * D_DIM * F_DIM + (size_t)d0 * F_DIM;
  const float* Wp1 = dec_w + (size_t)p1 * D_DIM * F_DIM + (size_t)d0 * F_DIM;
  const float* Wp2 = dec_w + (size_t)p2 * D_DIM * F_DIM + (size_t)d0 * F_DIM;

  const int steps_per_src = F_DIM / TKC;            // 256
  const int nsteps        = nsrc * steps_per_src;   // up to 768

  StageRegs r;
  prefetch_chunk(r, Ap0, Wp0, F_DIM, F_DIM, 0);
  commit_chunk(r, sA[0], sB[0]);
  __syncthreads();

  for (int s = 0; s < nsteps; ++s) {
    const int cur = s & 1;
    if (s + 1 < nsteps) {
      const int sn = s + 1;
      const int j  = sn >> 8;                       // source index 0..2
      const int k0 = (sn & (steps_per_src - 1)) * TKC;
      const float* Aj = (j == 0) ? Ap0 : (j == 1) ? Ap1 : Ap2;
      const float* Wj = (j == 0) ? Wp0 : (j == 1) ? Wp1 : Wp2;
      prefetch_chunk(r, Aj, Wj, F_DIM, F_DIM, k0);
    }
    mma_step(sA[cur], sB[cur], acc, wm, wn);
    if (s + 1 < nsteps)
      commit_chunk(r, sA[cur ^ 1], sB[cur ^ 1]);
    __syncthreads();
  }

  float* out = recons + (size_t)t * B_TOK * D_DIM;
  const int nlane = lane & 15;
  const int mbase = (lane >> 4) << 3;
#pragma unroll
  for (int ni = 0; ni < 4; ++ni) {
    const int dg = d0 + wn * 64 + ni * 16 + nlane;
#pragma unroll
    for (int mi = 0; mi < 4; ++mi) {
      const int mg0 = b0 + wm * 64 + mi * 16 + mbase;
#pragma unroll
      for (int i = 0; i < 8; ++i)
        out[(size_t)(mg0 + i) * D_DIM + dg] = acc[mi][ni][i];
    }
  }
}

// ---------------------------------------------------------------------------
// Losses
// ---------------------------------------------------------------------------
__global__ __launch_bounds__(256) void clt_sqerr(
    const float* __restrict__ recons, const float* __restrict__ tgt,
    float* __restrict__ ws_sq) {
  __shared__ float red[256];
  const int N = L_LAYERS * B_TOK * D_DIM;
  float s = 0.f;
  for (int idx = blockIdx.x * 256 + threadIdx.x; idx < N; idx += gridDim.x * 256) {
    float d = recons[idx] - tgt[idx];
    s += d * d;
  }
  float tot = block_reduce(s, red);
  if (threadIdx.x == 0) atomicAdd(ws_sq, tot);
}

__global__ void clt_init_ws(float* ws) {
  if (threadIdx.x < 16) ws[threadIdx.x] = 0.f;
}

// ws layout: [0]=sq_sum, [1]=l1_sum, [2..9]=l0 counts per layer
__global__ void clt_finalize(const float* __restrict__ ws,
                             float* __restrict__ scal) {
  if (threadIdx.x == 0) {
    const float recon = ws[0] * (1.0f / (float)(L_LAYERS * B_TOK * D_DIM));
    const float spars = 1e-4f * ws[1] * (1.0f / (float)(L_LAYERS * B_TOK * F_DIM));
    scal[0] = recon + spars;  // loss
    scal[1] = recon;          // recon_loss
    scal[2] = spars;          // sparsity_loss
  }
  if (threadIdx.x < L_LAYERS)
    scal[3 + threadIdx.x] = ws[2 + threadIdx.x] * (1.0f / (float)B_TOK);
}

// ---------------------------------------------------------------------------
extern "C" void kernel_launch(void* const* d_in, const int* in_sizes, int n_in,
                              void* d_out, int out_size, void* d_ws, size_t ws_size,
                              hipStream_t stream) {
  const float* res   = (const float*)d_in[0];  // [L, B, D]
  const float* tgt   = (const float*)d_in[1];  // [L, B, D]
  const float* enc_w = (const float*)d_in[2];  // [L, F, D]
  const float* enc_b = (const float*)d_in[3];  // [L, F]
  const float* dec_w = (const float*)d_in[4];  // [P, D, F]

  float* out    = (float*)d_out;
  float* recons = out;                                          // L*B*D
  float* feats  = out + (size_t)L_LAYERS * B_TOK * D_DIM;       // L*B*F
  float* scal   = feats + (size_t)L_LAYERS * B_TOK * F_DIM;     // 3 + L scalars
  float* ws     = (float*)d_ws;

  clt_init_ws<<<1, 32, 0, stream>>>(ws);

  dim3 eg(F_DIM / TNB, B_TOK / TMB, L_LAYERS);   // 64 x 4 x 8
  clt_encoder<<<eg, 256, 0, stream>>>(res, enc_w, enc_b, feats, &ws[1], &ws[2]);

  dim3 dg(D_DIM / TNB, B_TOK / TMB, L_LAYERS);   // 8 x 4 x 8
  clt_decoder<<<dg, 256, 0, stream>>>(feats, dec_w, recons);

  clt_sqerr<<<2048, 256, 0, stream>>>(recons, tgt, &ws[0]);
  clt_finalize<<<1, 32, 0, stream>>>(ws, scal);
}
